// AttnBlockVAE_30391188587275
// MI455X (gfx1250) — matrix operand order repaired and checked by reading, not compiled, not run on hardware
//
#include <hip/hip_runtime.h>
#include <hip/hip_bf16.h>
#include <math.h>

#define GROUPS 32
#define EPS 1e-5f

typedef __attribute__((ext_vector_type(16))) __bf16 v16bf;
typedef __attribute__((ext_vector_type(8)))  float  v8f;

union FragAB { v16bf v; uint4 q[2]; };

__device__ __forceinline__ unsigned short f2bf(float f) {
  unsigned u = __float_as_uint(f);
  u += 0x7FFFu + ((u >> 16) & 1u);          // round-to-nearest-even
  return (unsigned short)(u >> 16);
}

// --------- async global->LDS (CDNA5 GLOBAL_LOAD_ASYNC_TO_LDS_B128, ASYNCcnt) ---------
#if __has_builtin(__builtin_amdgcn_global_load_async_to_lds_b128)
#define HAS_ASYNC_LDS 1
#else
#define HAS_ASYNC_LDS 0
#endif

#if HAS_ASYNC_LDS
typedef int v4i_vec __attribute__((vector_size(16)));
typedef __attribute__((address_space(1))) v4i_vec* g_v4i_ptr;   // global int4*
typedef __attribute__((address_space(3))) v4i_vec* l_v4i_ptr;   // LDS int4*
#endif

// copy 32 contiguous bytes (16 bf16) global -> LDS, per thread
__device__ __forceinline__ void copy32_g2l(const unsigned short* src, unsigned short* dst_lds) {
#if HAS_ASYNC_LDS
  // the immediate offset is added to both the global and the LDS address
  __builtin_amdgcn_global_load_async_to_lds_b128(
      (g_v4i_ptr)src, (l_v4i_ptr)dst_lds, 0, 0);
  __builtin_amdgcn_global_load_async_to_lds_b128(
      (g_v4i_ptr)src, (l_v4i_ptr)dst_lds, 16, 0);
#else
  uint4 d0 = ((const uint4*)src)[0];
  uint4 d1 = ((const uint4*)src)[1];
  *(uint4*)dst_lds       = d0;
  *(uint4*)(dst_lds + 8) = d1;
#endif
}

__device__ __forceinline__ void wait_async_lds() {
#if HAS_ASYNC_LDS
#if __has_builtin(__builtin_amdgcn_s_wait_asynccnt)
  __builtin_amdgcn_s_wait_asynccnt(0);
#else
  asm volatile("s_wait_asynccnt 0x0" ::: "memory");
#endif
#endif
}

// ---------------- tiled bf16 WMMA GEMM ----------------
// C[M,N] = A[M,K] * B[K,N] (+ epilogue), bf16 inputs, fp32 accumulate.
// TA: A stored [K,M] (read transposed).  TB: B stored [N,K] (read transposed).
// MODE: 0 = +bias -> bf16 out (QKV)
//       1 = *scale -> f32 out (scores)
//       2 = bf16 out (P*V)
//       3 = +bias +residual -> f32 out (proj)
#define BMT 128
#define BNT 128
#define BKT 32
#define BKP 40   // padded LDS row stride (ushort units); 80B rows, 16B-aligned

template<int MODE, bool TA, bool TB>
__global__ void __launch_bounds__(256)
gemm_wmma_bf16(const unsigned short* __restrict__ A, long long strideAb, int lda,
               const unsigned short* __restrict__ B, long long strideBb, int ldb,
               void* __restrict__ Cv, long long strideCb, int ldc,
               const float* __restrict__ bias,
               const float* __restrict__ res, long long strideRb,
               float scale, int K)
{
  __shared__ __align__(16) unsigned short As[2][BMT * BKP];  // As[buf][m][k]
  __shared__ __align__(16) unsigned short Bs[2][BNT * BKP];  // Bs[buf][n][k]

  const int tid  = threadIdx.x;
  const int lane = tid & 31;
  const int wid  = tid >> 5;
  const int g    = (lane >> 4) & 1;   // lane half
  const int lm   = lane & 15;
  const int mwave = wid & 1;          // 2 waves along M (64 rows each)
  const int nwave = wid >> 1;         // 4 waves along N (32 cols each)

  const int m0 = blockIdx.y * BMT;
  const int n0 = blockIdx.x * BNT;
  const int z  = blockIdx.z;

  const unsigned short* Ab = A + (long long)z * strideAb;
  const unsigned short* Bb = B + (long long)z * strideBb;

  // ---- tile staging (contiguous paths use async-to-LDS; transposed scatter stays sync)
  auto stageA = [&](int k0, int buf) {
    if (!TA) {
      const int m = tid >> 1, kb = (tid & 1) * 16;
      const unsigned short* src = Ab + (long long)(m0 + m) * lda + (k0 + kb);
      copy32_g2l(src, &As[buf][m * BKP + kb]);
    } else {  // A stored [K, M]: coalesced along M, scatter into LDS
      const int k = tid >> 3, mb = (tid & 7) * 16;
      const unsigned short* src = Ab + (long long)(k0 + k) * lda + (m0 + mb);
      uint4 d0 = ((const uint4*)src)[0];
      uint4 d1 = ((const uint4*)src)[1];
      unsigned short tmp[16];
      *(uint4*)&tmp[0] = d0; *(uint4*)&tmp[8] = d1;
#pragma unroll
      for (int i = 0; i < 16; ++i) As[buf][(mb + i) * BKP + k] = tmp[i];
    }
  };
  auto stageB = [&](int k0, int buf) {
    if (!TB) {  // B stored [K, N]: coalesced along N, scatter into LDS
      const int k = tid >> 3, nb = (tid & 7) * 16;
      const unsigned short* src = Bb + (long long)(k0 + k) * ldb + (n0 + nb);
      uint4 d0 = ((const uint4*)src)[0];
      uint4 d1 = ((const uint4*)src)[1];
      unsigned short tmp[16];
      *(uint4*)&tmp[0] = d0; *(uint4*)&tmp[8] = d1;
#pragma unroll
      for (int i = 0; i < 16; ++i) Bs[buf][(nb + i) * BKP + k] = tmp[i];
    } else {    // B stored [N, K]: contiguous both ways
      const int n = tid >> 1, kb = (tid & 1) * 16;
      const unsigned short* src = Bb + (long long)(n0 + n) * ldb + (k0 + kb);
      copy32_g2l(src, &Bs[buf][n * BKP + kb]);
    }
  };

  v8f acc[4][2];
#pragma unroll
  for (int mt = 0; mt < 4; ++mt)
#pragma unroll
    for (int nt = 0; nt < 2; ++nt)
#pragma unroll
      for (int v = 0; v < 8; ++v) acc[mt][nt][v] = 0.0f;

  // preload tile 0
  stageA(0, 0);
  stageB(0, 0);
  int cur = 0;

  for (int k0 = 0; k0 < K; k0 += BKT) {
    wait_async_lds();     // my async writes to buf[cur] are in LDS
    __syncthreads();      // everyone's writes to buf[cur] visible; buf[cur^1] reads done
    if (k0 + BKT < K) {   // overlap next-tile staging with WMMA on current tile
      stageA(k0 + BKT, cur ^ 1);
      stageB(k0 + BKT, cur ^ 1);
    }

    // ---- fragments per CDNA5 ISA layouts (two ds_load_b128 each)
    // A 16x32: lane(g,lm): VGPR0-3 K=8g+0..7, VGPR4-7 K=16+8g+0..7, row M=lm
    // B 32x16: lane(g,lm): VGPR0-3 K=8g+0..7, VGPR4-7 K=16+8g+0..7, col N=lm
    const unsigned short* as = As[cur];
    const unsigned short* bs = Bs[cur];
    FragAB af[4], bf[2];
#pragma unroll
    for (int mt = 0; mt < 4; ++mt) {
      const int r = (mwave * 64 + mt * 16 + lm) * BKP;
      af[mt].q[0] = *(const uint4*)&as[r + 8 * g];
      af[mt].q[1] = *(const uint4*)&as[r + 16 + 8 * g];
    }
#pragma unroll
    for (int nt = 0; nt < 2; ++nt) {
      const int r = (nwave * 32 + nt * 16 + lm) * BKP;
      bf[nt].q[0] = *(const uint4*)&bs[r + 8 * g];
      bf[nt].q[1] = *(const uint4*)&bs[r + 16 + 8 * g];
    }
#pragma unroll
    for (int mt = 0; mt < 4; ++mt)
#pragma unroll
      for (int nt = 0; nt < 2; ++nt)
        acc[mt][nt] = __builtin_amdgcn_wmma_f32_16x16x32_bf16(
            false, af[mt].v, false, bf[nt].v, (short)0, acc[mt][nt], false, false);

    cur ^= 1;
  }

  // ---- epilogue: D layout -> VGPR v, lane(g,lm): M = v + 8g, N = lm
#pragma unroll
  for (int mt = 0; mt < 4; ++mt) {
    const int gmBase = m0 + mwave * 64 + mt * 16 + 8 * g;
#pragma unroll
    for (int nt = 0; nt < 2; ++nt) {
      const int gn = n0 + nwave * 32 + nt * 16 + lm;
#pragma unroll
      for (int v = 0; v < 8; ++v) {
        const int gm = gmBase + v;
        float val = acc[mt][nt][v];
        const long long idx = (long long)gm * ldc + gn;
        if (MODE == 0) {
          val += bias[gm];
          ((unsigned short*)Cv)[(long long)z * strideCb + idx] = f2bf(val);
        } else if (MODE == 1) {
          ((float*)Cv)[(long long)z * strideCb + idx] = val * scale;
        } else if (MODE == 2) {
          ((unsigned short*)Cv)[(long long)z * strideCb + idx] = f2bf(val);
        } else {
          val += bias[gm] + res[(long long)z * strideRb + idx];
          ((float*)Cv)[(long long)z * strideCb + idx] = val;
        }
      }
    }
  }
}

// ---------------- GroupNorm ----------------
__global__ void __launch_bounds__(256)
gn_stats(const float* __restrict__ x, float* __restrict__ stats, int chunk)
{
  __shared__ float red[256];
  const int bg = blockIdx.x;                 // b*GROUPS + g (groups contiguous in C)
  const float* p = x + (long long)bg * chunk;
  float s = 0.f, ss = 0.f;
  for (int i = threadIdx.x; i < chunk; i += 256) { float v = p[i]; s += v; ss += v * v; }
  red[threadIdx.x] = s; __syncthreads();
  for (int t = 128; t > 0; t >>= 1) { if (threadIdx.x < t) red[threadIdx.x] += red[threadIdx.x + t]; __syncthreads(); }
  const float sum = red[0]; __syncthreads();
  red[threadIdx.x] = ss; __syncthreads();
  for (int t = 128; t > 0; t >>= 1) { if (threadIdx.x < t) red[threadIdx.x] += red[threadIdx.x + t]; __syncthreads(); }
  if (threadIdx.x == 0) {
    const float mean = sum / (float)chunk;
    const float var  = red[0] / (float)chunk - mean * mean;
    stats[2 * bg]     = mean;
    stats[2 * bg + 1] = rsqrtf(var + EPS);
  }
}

__global__ void __launch_bounds__(256)
gn_apply(const float* __restrict__ x, const float* __restrict__ stats,
         const float* __restrict__ w, const float* __restrict__ bprm,
         unsigned short* __restrict__ hb, int C, int HW)
{
  const long long idx = (long long)blockIdx.x * 256 + threadIdx.x;
  const int c = (int)((idx / HW) % C);
  const int b = (int)(idx / ((long long)C * HW));
  const int sg = b * GROUPS + (c >> 4);      // C/GROUPS == 16
  const float mean = stats[2 * sg], rstd = stats[2 * sg + 1];
  hb[idx] = f2bf((x[idx] - mean) * rstd * w[c] + bprm[c]);
}

__global__ void __launch_bounds__(256)
f32_to_bf16(const float* __restrict__ src, unsigned short* __restrict__ dst, int n)
{
  const int i = blockIdx.x * 256 + threadIdx.x;
  if (i < n) dst[i] = f2bf(src[i]);
}

// ---------------- row softmax: fp32 in, bf16 out in-place (front half of row) --
__global__ void __launch_bounds__(256)
softmax_row(float* __restrict__ S, int N)
{
  __shared__ float red[256];
  const long long row = blockIdx.x;
  float* r = S + row * N;
  unsigned short* p = (unsigned short*)S + row * (2LL * N);  // ld = 2N bf16
  float e[16];
  float m = -3.402823466e+38f;
#pragma unroll
  for (int i = 0; i < 16; ++i) m = fmaxf(m, r[threadIdx.x + (i << 8)]);
  red[threadIdx.x] = m; __syncthreads();
  for (int t = 128; t > 0; t >>= 1) { if (threadIdx.x < t) red[threadIdx.x] = fmaxf(red[threadIdx.x], red[threadIdx.x + t]); __syncthreads(); }
  const float rowmax = red[0]; __syncthreads();
  float s = 0.f;
#pragma unroll
  for (int i = 0; i < 16; ++i) { e[i] = __expf(r[threadIdx.x + (i << 8)] - rowmax); s += e[i]; }
  red[threadIdx.x] = s; __syncthreads();
  for (int t = 128; t > 0; t >>= 1) { if (threadIdx.x < t) red[threadIdx.x] += red[threadIdx.x + t]; __syncthreads(); }
  const float inv = 1.0f / red[0];
  __syncthreads();                           // all fp32 reads done before bf16 overwrite
#pragma unroll
  for (int i = 0; i < 16; ++i) p[threadIdx.x + (i << 8)] = f2bf(e[i] * inv);
}

// ---------------- host orchestration ----------------
extern "C" void kernel_launch(void* const* d_in, const int* in_sizes, int n_in,
                              void* d_out, int out_size, void* d_ws, size_t ws_size,
                              hipStream_t stream) {
  const int Bn = 4, C = 512, HW = 4096;
  const long long CHW = (long long)C * HW;

  const float* x      = (const float*)d_in[0];
  const float* norm_w = (const float*)d_in[1];
  const float* norm_b = (const float*)d_in[2];
  const float* q_w    = (const float*)d_in[3];
  const float* q_b    = (const float*)d_in[4];
  const float* k_w    = (const float*)d_in[5];
  const float* k_b    = (const float*)d_in[6];
  const float* v_w    = (const float*)d_in[7];
  const float* v_b    = (const float*)d_in[8];
  const float* p_w    = (const float*)d_in[9];
  const float* p_b    = (const float*)d_in[10];

  // workspace carve-up (~136 MB total)
  char* ws = (char*)d_ws;
  auto carve = [&](size_t bytes) { char* p = ws; ws += (bytes + 255) & ~(size_t)255; return p; };
  unsigned short* wq = (unsigned short*)carve((size_t)C * C * 2);
  unsigned short* wk = (unsigned short*)carve((size_t)C * C * 2);
  unsigned short* wv = (unsigned short*)carve((size_t)C * C * 2);
  unsigned short* wp = (unsigned short*)carve((size_t)C * C * 2);
  float* stats       = (float*)carve((size_t)Bn * GROUPS * 2 * 4);
  unsigned short* hb = (unsigned short*)carve((size_t)Bn * CHW * 2);  // h, reused as O
  unsigned short* qb = (unsigned short*)carve((size_t)Bn * CHW * 2);
  unsigned short* kb = (unsigned short*)carve((size_t)Bn * CHW * 2);
  unsigned short* vb = (unsigned short*)carve((size_t)Bn * CHW * 2);
  float* S           = (float*)carve((size_t)HW * HW * 4);            // per-batch scores

  const int WN = C * C;
  f32_to_bf16<<<(WN + 255) / 256, 256, 0, stream>>>(q_w, wq, WN);
  f32_to_bf16<<<(WN + 255) / 256, 256, 0, stream>>>(k_w, wk, WN);
  f32_to_bf16<<<(WN + 255) / 256, 256, 0, stream>>>(v_w, wv, WN);
  f32_to_bf16<<<(WN + 255) / 256, 256, 0, stream>>>(p_w, wp, WN);

  gn_stats<<<Bn * GROUPS, 256, 0, stream>>>(x, stats, (C / GROUPS) * HW);
  gn_apply<<<(int)((Bn * CHW) / 256), 256, 0, stream>>>(x, stats, norm_w, norm_b, hb, C, HW);

  // QKV: [512 x 4096] = W[512,512] * h[512,4096] per batch
  dim3 gQKV(HW / BNT, C / BMT, Bn);
  gemm_wmma_bf16<0, false, false><<<gQKV, 256, 0, stream>>>(wq, 0, C, hb, CHW, HW, qb, CHW, HW, q_b, nullptr, 0, 1.f, C);
  gemm_wmma_bf16<0, false, false><<<gQKV, 256, 0, stream>>>(wk, 0, C, hb, CHW, HW, kb, CHW, HW, k_b, nullptr, 0, 1.f, C);
  gemm_wmma_bf16<0, false, false><<<gQKV, 256, 0, stream>>>(wv, 0, C, hb, CHW, HW, vb, CHW, HW, v_b, nullptr, 0, 1.f, C);

  const float scl = 1.0f / sqrtf((float)C);
  for (int b = 0; b < Bn; ++b) {
    // S[t,s] = scale * sum_c Q[c,t] K[c,s]  (A transposed)
    dim3 gS(HW / BNT, HW / BMT, 1);
    gemm_wmma_bf16<1, true, false><<<gS, 256, 0, stream>>>(
        qb + b * CHW, 0, HW, kb + b * CHW, 0, HW, S, 0, HW, nullptr, nullptr, 0, scl, C);
    softmax_row<<<HW, 256, 0, stream>>>(S, HW);
    // O[c,t] = sum_s V[c,s] P[t,s]  (B transposed, P ld = 2*HW bf16) -> overwrite hb
    dim3 gO(HW / BNT, C / BMT, 1);
    gemm_wmma_bf16<2, false, true><<<gO, 256, 0, stream>>>(
        vb + b * CHW, 0, HW, (unsigned short*)S, 0, 2 * HW, hb + b * CHW, 0, HW,
        nullptr, nullptr, 0, 1.f, HW);
  }

  // proj + bias + residual -> fp32 out
  dim3 gP(HW / BNT, C / BMT, Bn);
  gemm_wmma_bf16<3, false, false><<<gP, 256, 0, stream>>>(
      wp, 0, C, hb, CHW, HW, (float*)d_out, CHW, HW, p_b, x, CHW, 1.f, C);
}
